// PointTransformerClassification_56538949485048
// MI455X (gfx1250) — compile-verified, matrix operand order checked by reading, not run on hardware
//
#include <hip/hip_runtime.h>
#include <hip/hip_bf16.h>

// Point Transformer classification forward pass for gfx1250 (MI455X).
// All matmuls go through V_WMMA_F32_16X16X32_F16 (f32 accumulate).
// GEMM is templated on <KC, NC, RELU, RES, BIAS> so K-chunk loops fully
// unroll (dense back-to-back WMMA), bounds checks and epilogue flags
// constant-fold, A loads vectorize to b128, and the epilogue transposes
// through per-wave LDS so outputs go out as b128 stores.

typedef __attribute__((ext_vector_type(16))) _Float16 v16h;
typedef __attribute__((ext_vector_type(8)))  float    v8f;

#define TB 256
#define WPB 8   // waves per block (wave32)
#define MT  2   // m-tiles per wave

// ---------------------------------------------------------------------------
// Fused GEMM:  C[M,NC] = act( A[M,KC] * W[KC,NC] + bias + residual )
// row-major f32 in/out, f16 WMMA tiles, K padded to 32 with zeros.
// grid.x covers M-tiles (WPB*MT per block), grid.y = N-tiles.
// ---------------------------------------------------------------------------
template <int KC, int NC, bool RELU, bool RES, bool BIAS>
__launch_bounds__(TB)
__global__ void gemm_wmma_kernel(const float* __restrict__ A,
                                 const float* __restrict__ W,
                                 const float* __restrict__ bias,
                                 const float* __restrict__ residual,
                                 float* __restrict__ C,
                                 int M)
{
    constexpr int NCHUNK = (KC + 31) / 32;

    __shared__ _Float16 Wl[16 * 32];              // [n][k] transposed W chunk
    __shared__ float    Ct[WPB][MT][16][16];      // per-wave transpose buffer

    const int lane = threadIdx.x & 31;
    const int wid  = threadIdx.x >> 5;
    const int Mtiles = (M + 15) >> 4;
    const int t0 = (blockIdx.x * WPB + wid) * MT;       // first m-tile of wave
    const int tileN = blockIdx.y;

    const int nl = lane & 15;           // column within tile (B/C/D layouts)
    const int mh = (lane >> 4) << 3;    // K-base offset (A/B) == M offset (C/D)
    const int gn = tileN * 16 + nl;

    bool  tv[MT];
    const float* Arow[MT];
#pragma unroll
    for (int t = 0; t < MT; ++t) {
        tv[t] = (t0 + t) < Mtiles;
        int tm = tv[t] ? (t0 + t) : 0;
        Arow[t] = A + (size_t)(tm * 16 + nl) * (size_t)KC;
    }

    float bv = 0.0f;
    if constexpr (BIAS) {
        int gnc = (gn < NC) ? gn : 0;
        float b0 = bias[gnc];
        bv = (gn < NC) ? b0 : 0.0f;
    }
    v8f acc[MT];
#pragma unroll
    for (int t = 0; t < MT; ++t)
#pragma unroll
        for (int v = 0; v < 8; ++v) acc[t][v] = bv;

#pragma unroll
    for (int c0 = 0; c0 < NCHUNK; ++c0) {
        const int k0 = c0 * 32;

        // cooperative, branchless staging of W[k0:k0+32, tileN*16:+16] -> LDS
#pragma unroll
        for (int half = 0; half < 2; ++half) {
            int e   = (int)threadIdx.x + half * TB;     // 0..511
            int kk  = e >> 4;                           // 0..31
            int nn  = e & 15;
            int gk  = k0 + kk;
            int gnn = tileN * 16 + nn;
            int gks  = (gk < KC) ? gk : 0;
            int gnns = (gnn < NC) ? gnn : 0;
            float w = W[(size_t)gks * NC + gnns];
            w = (gk < KC && gnn < NC) ? w : 0.0f;
            Wl[nn * 32 + kk] = (_Float16)w;
        }
        __syncthreads();

        if constexpr (KC > 32) {
            if (k0 + 32 < KC) {
#pragma unroll
                for (int t = 0; t < MT; ++t)
                    __builtin_prefetch(Arow[t] + k0 + 32, 0, 1);
            }
        }

        // B fragment from LDS: two contiguous 16B runs per lane
        v16h bfr;
#pragma unroll
        for (int i = 0; i < 8; ++i) bfr[i]     = Wl[nl * 32 + mh + i];
#pragma unroll
        for (int i = 0; i < 8; ++i) bfr[8 + i] = Wl[nl * 32 + mh + 16 + i];

#pragma unroll
        for (int t = 0; t < MT; ++t) {
            v16h a;
#pragma unroll
            for (int i = 0; i < 16; ++i) {
                const int kk = k0 + mh + (i < 8 ? i : i + 8);
                if constexpr (KC % 32 == 0) {
                    a[i] = (_Float16)Arow[t][kk];           // always in range
                } else if constexpr (KC == 16) {
                    // valid iff i < 8 (mh<=8 -> kk<=15); rest statically zero
                    a[i] = (i < 8) ? (_Float16)Arow[t][kk] : (_Float16)0.0f;
                } else {
                    int kc = (kk < KC) ? kk : 0;            // clamped, branchless
                    float av = Arow[t][kc];
                    a[i] = (kk < KC) ? (_Float16)av : (_Float16)0.0f;
                }
            }
            acc[t] = __builtin_amdgcn_wmma_f32_16x16x32_f16(false, a, false, bfr,
                                                            (short)0, acc[t],
                                                            false, false);
        }
        __syncthreads();
    }

    // -------- epilogue: transpose via per-wave LDS, then b128 row stores ----
#pragma unroll
    for (int t = 0; t < MT; ++t)
#pragma unroll
        for (int v = 0; v < 8; ++v)
            Ct[wid][t][mh + v][nl] = acc[t][v];
    // same-wave DS ops are in-order; compiler inserts the dscnt wait.

    const int row_l = nl;               // local row this lane stores
    const int colh  = (lane >> 4) * 8;  // 0 or 8: column half

#pragma unroll
    for (int t = 0; t < MT; ++t) {
        if (!tv[t]) continue;           // wave-uniform
        const int grow = (t0 + t) * 16 + row_l;
        const float* lrow = &Ct[wid][t][row_l][colh];
        float* crow = C + (size_t)grow * NC + tileN * 16 + colh;

        if constexpr (NC % 16 == 0) {
            float4 o0 = *(const float4*)(lrow);
            float4 o1 = *(const float4*)(lrow + 4);
            if constexpr (RES) {
                const float* rrow = residual + (size_t)grow * NC + tileN * 16 + colh;
                float4 r0 = *(const float4*)(rrow);
                float4 r1 = *(const float4*)(rrow + 4);
                o0.x += r0.x; o0.y += r0.y; o0.z += r0.z; o0.w += r0.w;
                o1.x += r1.x; o1.y += r1.y; o1.z += r1.z; o1.w += r1.w;
            }
            if constexpr (RELU) {
                o0.x = fmaxf(o0.x, 0.0f); o0.y = fmaxf(o0.y, 0.0f);
                o0.z = fmaxf(o0.z, 0.0f); o0.w = fmaxf(o0.w, 0.0f);
                o1.x = fmaxf(o1.x, 0.0f); o1.y = fmaxf(o1.y, 0.0f);
                o1.z = fmaxf(o1.z, 0.0f); o1.w = fmaxf(o1.w, 0.0f);
            }
            *(float4*)(crow)     = o0;
            *(float4*)(crow + 4) = o1;
        } else {
            // partial N tile (NC == 40 case): scalar guarded stores
#pragma unroll
            for (int i = 0; i < 8; ++i) {
                int col = tileN * 16 + colh + i;
                if (col < NC) {
                    float o = lrow[i];
                    if constexpr (RES) o += residual[(size_t)grow * NC + col];
                    if constexpr (RELU) o = fmaxf(o, 0.0f);
                    C[(size_t)grow * NC + col] = o;
                }
            }
        }
    }
}

// ---------------------------------------------------------------------------
// Brute-force KNN (k=16): one thread per query, register insertion sort.
// ---------------------------------------------------------------------------
__global__ void knn_kernel(const float* __restrict__ q,
                           const float* __restrict__ p,
                           int Mq, int Np, int* __restrict__ idx, int B)
{
    int tid = blockIdx.x * blockDim.x + threadIdx.x;
    if (tid >= B * Mq) return;
    int b = tid / Mq;
    const float* qp = q + (size_t)tid * 3;
    float qx = qp[0], qy = qp[1], qz = qp[2];
    const float* pb = p + (size_t)b * Np * 3;

    float bd[16]; int bi[16];
#pragma unroll
    for (int t = 0; t < 16; ++t) { bd[t] = 3.0e38f; bi[t] = 0; }

    for (int j = 0; j < Np; ++j) {
        float dx = qx - pb[j * 3 + 0];
        float dy = qy - pb[j * 3 + 1];
        float dz = qz - pb[j * 3 + 2];
        float d = dx * dx + dy * dy + dz * dz;
        if (d < bd[15]) {
            int pos = 0;
#pragma unroll
            for (int t = 0; t < 16; ++t) pos += (bd[t] <= d) ? 1 : 0;
#pragma unroll
            for (int t = 15; t > 0; --t) {
                if (t > pos) { bd[t] = bd[t - 1]; bi[t] = bi[t - 1]; }
            }
#pragma unroll
            for (int t = 0; t < 16; ++t) {
                if (t == pos) { bd[t] = d; bi[t] = j; }
            }
        }
    }
    int* o = idx + (size_t)tid * 16;
#pragma unroll
    for (int t = 0; t < 16; ++t) o[t] = bi[t];
}

// ---------------------------------------------------------------------------
// Farthest point sampling: one block per batch, LDS distance array + argmax.
// ---------------------------------------------------------------------------
__global__ void fps_kernel(const float* __restrict__ c, int Np, int m,
                           int* __restrict__ fidx)
{
    __shared__ float sdist[1024];
    __shared__ float rv[TB];
    __shared__ int   ri[TB];
    __shared__ float lastp[3];

    int b = blockIdx.x;
    int t = threadIdx.x;
    const float* cb = c + (size_t)b * Np * 3;

    for (int i = t; i < Np; i += TB) sdist[i] = 1.0e10f;
    if (t == 0) {
        fidx[b * m] = 0;
        lastp[0] = cb[0]; lastp[1] = cb[1]; lastp[2] = cb[2];
    }
    __syncthreads();

    for (int s = 1; s < m; ++s) {
        float lx = lastp[0], ly = lastp[1], lz = lastp[2];
        float bestd = -1.0f; int besti = 0;
        for (int i = t; i < Np; i += TB) {
            float dx = cb[i * 3 + 0] - lx;
            float dy = cb[i * 3 + 1] - ly;
            float dz = cb[i * 3 + 2] - lz;
            float d  = dx * dx + dy * dy + dz * dz;
            float nd = fminf(sdist[i], d);
            sdist[i] = nd;
            if (nd > bestd) { bestd = nd; besti = i; }
        }
        rv[t] = bestd; ri[t] = besti;
        __syncthreads();
        for (int st = TB / 2; st > 0; st >>= 1) {
            if (t < st && rv[t + st] > rv[t]) { rv[t] = rv[t + st]; ri[t] = ri[t + st]; }
            __syncthreads();
        }
        if (t == 0) {
            int bi = ri[0];
            fidx[b * m + s] = bi;
            lastp[0] = cb[bi * 3 + 0];
            lastp[1] = cb[bi * 3 + 1];
            lastp[2] = cb[bi * 3 + 2];
        }
        __syncthreads();
    }
}

// ---------------------------------------------------------------------------
// local[b,n,k,:] = coords[b, idx[b,n,k], :] - coords[b,n,:]
// ---------------------------------------------------------------------------
__global__ void build_local_kernel(const float* __restrict__ coords,
                                   const int* __restrict__ idx,
                                   float* __restrict__ local, int Nq, int B)
{
    long long tid = (long long)blockIdx.x * blockDim.x + threadIdx.x;
    long long total = (long long)B * Nq * 16 * 3;
    if (tid >= total) return;
    int d = (int)(tid % 3);
    long long t = tid / 3;
    int k = (int)(t % 16);
    long long bn = t / 16;
    int b = (int)(bn / Nq);
    int j = idx[bn * 16 + k];
    local[tid] = coords[((size_t)b * Nq + j) * 3 + d] - coords[(size_t)bn * 3 + d];
}

// ---------------------------------------------------------------------------
// g[b,n,k,c] = q[b,n,c] - kf[b, idx[b,n,k], c] + delta[b,n,k,c]
// ---------------------------------------------------------------------------
__global__ void g_build_kernel(const float* __restrict__ qf,
                               const float* __restrict__ kf,
                               const float* __restrict__ delta,
                               const int* __restrict__ idx,
                               float* __restrict__ g, int Nq, int B)
{
    long long tid = (long long)blockIdx.x * blockDim.x + threadIdx.x;
    long long total = (long long)B * Nq * 16 * 16;
    if (tid >= total) return;
    int c = (int)(tid & 15);
    long long t = tid >> 4;
    int k = (int)(t % 16);
    long long bn = t / 16;
    int b = (int)(bn / Nq);
    int j = idx[bn * 16 + k];
    g[tid] = qf[bn * 16 + c] - kf[((size_t)b * Nq + j) * 16 + c] + delta[tid];
}

// ---------------------------------------------------------------------------
// Softmax over the 16 neighbors + weighted sum of (v + delta).
// ---------------------------------------------------------------------------
__global__ void attn_kernel(const float* __restrict__ logits,
                            const float* __restrict__ vf,
                            const float* __restrict__ delta,
                            const int* __restrict__ idx,
                            float* __restrict__ out, int Nq, int B)
{
    long long tid = (long long)blockIdx.x * blockDim.x + threadIdx.x;
    long long total = (long long)B * Nq * 16;
    if (tid >= total) return;
    int c = (int)(tid & 15);
    long long bn = tid >> 4;
    int b = (int)(bn / Nq);

    float lv[16];
    float mx = -3.0e38f;
#pragma unroll
    for (int k = 0; k < 16; ++k) {
        lv[k] = logits[((size_t)bn * 16 + k) * 16 + c];
        mx = fmaxf(mx, lv[k]);
    }
    float s = 0.0f;
#pragma unroll
    for (int k = 0; k < 16; ++k) { lv[k] = __expf(lv[k] - mx); s += lv[k]; }
    float inv = 1.0f / s;
    float o = 0.0f;
#pragma unroll
    for (int k = 0; k < 16; ++k) {
        int j = idx[bn * 16 + k];
        o += lv[k] * inv *
             (vf[((size_t)b * Nq + j) * 16 + c] + delta[((size_t)bn * 16 + k) * 16 + c]);
    }
    out[tid] = o;
}

// ---------------------------------------------------------------------------
// dst[b, r, :] = src[b, idx[b, r], :]
// ---------------------------------------------------------------------------
__global__ void gather_rows_kernel(const float* __restrict__ src,
                                   const int* __restrict__ idx,
                                   float* __restrict__ dst,
                                   int rowsPerBatch, int C, int Nsrc, int B)
{
    long long tid = (long long)blockIdx.x * blockDim.x + threadIdx.x;
    long long total = (long long)B * rowsPerBatch * C;
    if (tid >= total) return;
    int c = (int)(tid % C);
    long long r = tid / C;
    int b = (int)(r / rowsPerBatch);
    int j = idx[r];
    dst[tid] = src[((size_t)b * Nsrc + j) * C + c];
}

// ---------------------------------------------------------------------------
// out[b,i,c] = max_k g[b,i,k,c]
// ---------------------------------------------------------------------------
__global__ void maxpool_k_kernel(const float* __restrict__ g,
                                 float* __restrict__ out, int m, int C, int B)
{
    long long tid = (long long)blockIdx.x * blockDim.x + threadIdx.x;
    long long total = (long long)B * m * C;
    if (tid >= total) return;
    int c = (int)(tid % C);
    long long bm = tid / C;
    float mx = -3.0e38f;
#pragma unroll
    for (int k = 0; k < 16; ++k)
        mx = fmaxf(mx, g[((size_t)bm * 16 + k) * C + c]);
    out[tid] = mx;
}

// ---------------------------------------------------------------------------
// out[b,c] = mean_i h[b,i,c]
// ---------------------------------------------------------------------------
__global__ void mean_kernel(const float* __restrict__ h,
                            float* __restrict__ out, int m, int C, int B)
{
    int tid = blockIdx.x * blockDim.x + threadIdx.x;
    if (tid >= B * C) return;
    int c = tid % C;
    int b = tid / C;
    float s = 0.0f;
    for (int i = 0; i < m; ++i) s += h[((size_t)b * m + i) * C + c];
    out[tid] = s * (1.0f / (float)m);
}

// ---------------------------------------------------------------------------
// Host side
// ---------------------------------------------------------------------------
template <int KC, int NC, bool RELU, bool RES = false, bool BIAS = true>
static inline void gemm(const float* A, const float* W, const float* bias,
                        const float* residual, float* C, int M, hipStream_t s)
{
    int Mtiles = (M + 15) / 16;
    dim3 grid((Mtiles + WPB * MT - 1) / (WPB * MT), (NC + 15) / 16);
    gemm_wmma_kernel<KC, NC, RELU, RES, BIAS>
        <<<grid, dim3(TB), 0, s>>>(A, W, bias, residual, C, M);
}

static inline dim3 g1d(long long n) { return dim3((unsigned)((n + TB - 1) / TB)); }

template <int CI, int CO>
static void enc_stage(const float* ccur, const float* hcur, int Ncur, int m,
                      const float* Wt, const float* Bb,
                      int* fidx, int* nidx, float* fc, float* grouped,
                      float* gout, float* hn, hipStream_t stream)
{
    const int B = 64, K = 16;
    fps_kernel<<<dim3(B), dim3(TB), 0, stream>>>(ccur, Ncur, m, fidx);
    gather_rows_kernel<<<g1d((long long)B * m * 3), dim3(TB), 0, stream>>>(
        ccur, fidx, fc, m, 3, Ncur, B);
    knn_kernel<<<g1d((long long)B * m), dim3(TB), 0, stream>>>(fc, ccur, m, Ncur, nidx, B);
    gather_rows_kernel<<<g1d((long long)B * m * K * CI), dim3(TB), 0, stream>>>(
        hcur, nidx, grouped, m * K, CI, Ncur, B);
    gemm<CI, CO, true>(grouped, Wt, Bb, nullptr, gout, B * m * K, stream);
    maxpool_k_kernel<<<g1d((long long)B * m * CO), dim3(TB), 0, stream>>>(
        gout, hn, m, CO, B);
}

extern "C" void kernel_launch(void* const* d_in, const int* in_sizes, int n_in,
                              void* d_out, int out_size, void* d_ws, size_t ws_size,
                              hipStream_t stream)
{
    (void)in_sizes; (void)n_in; (void)out_size; (void)ws_size;

    const int B = 64, N = 1024, K = 16;

    const float* x      = (const float*)d_in[0];
    const float* coords = (const float*)d_in[1];
    const float* in_w   = (const float*)d_in[2];
    const float* in_b   = (const float*)d_in[3];
    const float* lin1_w = (const float*)d_in[4];
    const float* lin1_b = (const float*)d_in[5];
    const float* q_w    = (const float*)d_in[6];
    const float* k_w    = (const float*)d_in[7];
    const float* v_w    = (const float*)d_in[8];
    const float* pos1_w = (const float*)d_in[9];
    const float* pos1_b = (const float*)d_in[10];
    const float* pos2_w = (const float*)d_in[11];
    const float* pos2_b = (const float*)d_in[12];
    const float* g1_w   = (const float*)d_in[13];
    const float* g1_b   = (const float*)d_in[14];
    const float* g2_w   = (const float*)d_in[15];
    const float* g2_b   = (const float*)d_in[16];
    const float* lin2_w = (const float*)d_in[17];
    const float* lin2_b = (const float*)d_in[18];
    const float* e0_w = (const float*)d_in[19];
    const float* e0_b = (const float*)d_in[20];
    const float* e1_w = (const float*)d_in[21];
    const float* e1_b = (const float*)d_in[22];
    const float* e2_w = (const float*)d_in[23];
    const float* e2_b = (const float*)d_in[24];
    const float* e3_w = (const float*)d_in[25];
    const float* e3_b = (const float*)d_in[26];
    const float* d1_w = (const float*)d_in[27];
    const float* d1_b = (const float*)d_in[28];
    const float* d2_w = (const float*)d_in[29];
    const float* d2_b = (const float*)d_in[30];
    const float* d3_w = (const float*)d_in[31];
    const float* d3_b = (const float*)d_in[32];

    float* ws = (float*)d_ws;
    size_t off = 0;
    auto alloc = [&](size_t n) { float* p = ws + off; off += n; return p; };

    // ---------------- input MLP + pt_block ----------------
    float* h0    = alloc((size_t)B * N * 32);
    int*   knn0  = (int*)alloc((size_t)B * N * K);
    float* y     = alloc((size_t)B * N * 16);
    float* qf    = alloc((size_t)B * N * 16);
    float* kf    = alloc((size_t)B * N * 16);
    float* vf    = alloc((size_t)B * N * 16);
    float* loc   = alloc((size_t)B * N * K * 3);
    float* tA    = alloc((size_t)B * N * K * 16);   // pos hidden / g / logits
    float* delta = alloc((size_t)B * N * K * 16);
    float* tB    = alloc((size_t)B * N * K * 16);   // relu(g@g1+b)
    float* att   = alloc((size_t)B * N * 16);
    float* h1    = alloc((size_t)B * N * 32);

    gemm<6, 32, true>(x, in_w, in_b, nullptr, h0, B * N, stream);

    knn_kernel<<<g1d((long long)B * N), dim3(TB), 0, stream>>>(coords, coords, N, N, knn0, B);

    gemm<32, 16, false>(h0, lin1_w, lin1_b, nullptr, y, B * N, stream);
    gemm<16, 16, false, false, false>(y, q_w, nullptr, nullptr, qf, B * N, stream);
    gemm<16, 16, false, false, false>(y, k_w, nullptr, nullptr, kf, B * N, stream);
    gemm<16, 16, false, false, false>(y, v_w, nullptr, nullptr, vf, B * N, stream);

    build_local_kernel<<<g1d((long long)B * N * K * 3), dim3(TB), 0, stream>>>(
        coords, knn0, loc, N, B);

    gemm<3, 16, true>(loc, pos1_w, pos1_b, nullptr, tA, B * N * K, stream);
    gemm<16, 16, false>(tA, pos2_w, pos2_b, nullptr, delta, B * N * K, stream);

    g_build_kernel<<<g1d((long long)B * N * K * 16), dim3(TB), 0, stream>>>(
        qf, kf, delta, knn0, tA, N, B);

    gemm<16, 16, true>(tA, g1_w, g1_b, nullptr, tB, B * N * K, stream);
    gemm<16, 16, false>(tB, g2_w, g2_b, nullptr, tA, B * N * K, stream);  // logits in tA

    attn_kernel<<<g1d((long long)B * N * 16), dim3(TB), 0, stream>>>(
        tA, vf, delta, knn0, att, N, B);

    gemm<16, 32, true, true>(att, lin2_w, lin2_b, h0 /*residual*/, h1, B * N, stream);

    // ---------------- encoder stages ----------------
    int*   fidx    = (int*)alloc((size_t)B * 256);
    int*   nidx    = (int*)alloc((size_t)B * 256 * K);
    float* fcA     = alloc((size_t)B * 256 * 3);
    float* fcB     = alloc((size_t)B * 256 * 3);
    float* grouped = alloc((size_t)B * 256 * K * 32);
    float* gout    = alloc((size_t)B * 256 * K * 64);
    float* hA      = alloc((size_t)B * 256 * 64);
    float* hB      = alloc((size_t)B * 256 * 64);

    enc_stage<32, 64>  (coords, h1, 1024, 256, e0_w, e0_b, fidx, nidx, fcA, grouped, gout, hA, stream);
    enc_stage<64, 128> (fcA,    hA,  256,  64, e1_w, e1_b, fidx, nidx, fcB, grouped, gout, hB, stream);
    enc_stage<128, 256>(fcB,    hB,   64,  16, e2_w, e2_b, fidx, nidx, fcA, grouped, gout, hA, stream);
    enc_stage<256, 512>(fcA,    hA,   16,   4, e3_w, e3_b, fidx, nidx, fcB, grouped, gout, hB, stream);

    // ---------------- head ----------------
    float* pooled = alloc((size_t)B * 512);
    float* y1     = alloc((size_t)B * 512);
    float* y2     = alloc((size_t)B * 256);

    mean_kernel<<<g1d((long long)B * 512), dim3(TB), 0, stream>>>(hB, pooled, 4, 512, B);

    gemm<512, 512, true>(pooled, d1_w, d1_b, nullptr, y1, B, stream);
    gemm<512, 256, true>(y1,     d2_w, d2_b, nullptr, y2, B, stream);
    gemm<256, 40, false>(y2,     d3_w, d3_b, nullptr, (float*)d_out, B, stream);
}
